// ConvDecoder_52312701665413
// MI455X (gfx1250) — matrix operand order repaired and checked by reading, not compiled, not run on hardware
//
#include <hip/hip_runtime.h>
#include <hip/hip_bf16.h>
#include <math.h>

// ConvS2S-style decoder for MI455X (gfx1250, wave32, WMMA).
// GEMMs -> v_wmma_f32_16x16x32_f16 with fragment-major pre-packed B.
// Conv weights stream through LDS via async-to-LDS double buffering (ASYNCcnt).

typedef __attribute__((ext_vector_type(16))) _Float16 v16h;
typedef __attribute__((ext_vector_type(8)))  _Float16 v8h;
typedef __attribute__((ext_vector_type(8)))  float    v8f;

#define Bb_ 8
#define Tt_ 512
#define Ss_ 1024
#define Ee_ 64
#define Vv_ 32000
static constexpr float SQ_ = 0.70710678118654752440f;

// ---- CDNA5 async-to-LDS primitives ----------------------------------------
#if __has_builtin(__builtin_amdgcn_s_wait_asynccnt)
#define WAIT_ASYNC(n) __builtin_amdgcn_s_wait_asynccnt((unsigned short)(n))
#else
#define WAIT_ASYNC(n) asm volatile("s_wait_asynccnt %0" ::"i"(n))
#endif

// GLOBAL_LOAD_ASYNC_TO_LDS_B128: copies global vaddr to LDS offset vdst, 16B per lane, ASYNCcnt.
__device__ __forceinline__ void async_load_b128(unsigned lds_off, const void* gptr) {
  asm volatile("global_load_async_to_lds_b128 %0, %1, off"
               :: "v"(lds_off), "v"(gptr) : "memory");
}

// ---------------- A fragment: row-major f16, ISA 7.12.2 16x32 layout --------
// lanes 0-15: M=lane, halves [0..7]=K+0..7, [8..15]=K+16..23 ; lanes 16-31: +8 on K
__device__ __forceinline__ v16h ld_a_rm(const _Float16* __restrict__ A,
                                        long lda, int m0, int k0, int lane) {
  const _Float16* p = A + (long)(m0 + (lane & 15)) * lda + k0 + ((lane & 16) ? 8 : 0);
  v16h r;
  ((v8h*)&r)[0] = *(const v8h*)(p);
  ((v8h*)&r)[1] = *(const v8h*)(p + 16);
  return r;
}

// ---------------- B pre-pack: f32 (strided) -> fragment-major f16 -----------
__global__ __launch_bounds__(32) void k_pack_b(
    const float* __restrict__ S, long sbz, long bk, long bn,
    _Float16* __restrict__ D, long obz, long o_nt, long o_kt) {
  const int lane = threadIdx.x;
  const long kt = blockIdx.x, nt = blockIdx.y, bz = blockIdx.z;
  const int n  = (int)nt * 16 + (lane & 15);
  const int kb = (int)kt * 32 + ((lane & 16) ? 8 : 0);
  const float* s = S + bz * sbz + (long)n * bn;
  v16h r;
#pragma unroll
  for (int h = 0; h < 16; ++h) r[h] = (_Float16)s[(long)(kb + (h < 8 ? h : h + 8)) * bk];
  *(v16h*)(D + ((bz * obz + nt * o_nt + kt * o_kt) * 32 + lane) * 16) = r;
}

// ---------------- generic WMMA GEMM:  C = act(scale*(A@Bpk) + bias) ---------
template <int NT>
__global__ __launch_bounds__(32) void k_wmma_gemm(
    const _Float16* __restrict__ A, long aBatch, long lda,
    const _Float16* __restrict__ Bpk, long bBatch, int kTiles,  // packed [nt][kt]
    const float* __restrict__ bias,
    float* __restrict__ C, long cBatch, int ldc,
    int K, int act, float scale) {
  const int lane = threadIdx.x;
  const long bz  = blockIdx.z;
  const _Float16* Ab = A + bz * aBatch;
  const _Float16* Bb = Bpk + bz * bBatch;
  float*          Cb = C + bz * cBatch;
  const int m0 = blockIdx.y * 16;
  const int nt0 = blockIdx.x * NT;

  v8f acc[NT];
#pragma unroll
  for (int j = 0; j < NT; ++j) acc[j] = (v8f){};

#pragma unroll 1
  for (int kt = 0, k0 = 0; k0 < K; k0 += 32, ++kt) {
    if (k0 + 32 < K)  // -> global_prefetch_b8
      __builtin_prefetch((const void*)(Ab + (long)(m0 + (lane & 15)) * lda + k0 + 32), 0, 3);
    v16h af = ld_a_rm(Ab, lda, m0, k0, lane);
    const _Float16* bp = Bb + (((long)nt0 * kTiles + kt) * 32 + lane) * 16;
#pragma unroll
    for (int j = 0; j < NT; ++j) {
      v16h bf = *(const v16h*)(bp + (long)j * kTiles * 512);
      acc[j] = __builtin_amdgcn_wmma_f32_16x16x32_f16(false, af, false, bf,
                                                      (short)0, acc[j], false, false);
    }
  }
  const int nlo = lane & 15;
  const int mo  = (lane & 16) ? 8 : 0;
#pragma unroll
  for (int j = 0; j < NT; ++j) {
    const int nc = (nt0 + j) * 16 + nlo;
    const float bv = bias ? bias[nc] : 0.f;
#pragma unroll
    for (int r = 0; r < 8; ++r) {
      float vv = acc[j][r] * scale + bv;
      if (act) vv = fmaxf(vv, 0.f);
      Cb[(long)(m0 + r + mo) * ldc + nc] = vv;
    }
  }
}

// ---------------- conv as sliding-window WMMA GEMM --------------------------
// out[b,t,o] = sum_{db,dt,i} y[b+db-2, t+dt-31, i] * W[db,dt,i,o] + bias[o]
// 640 chunks of K=32 (5 taps x 128); packed weights (8KB/chunk) stream into an
// LDS double buffer via global_load_async_to_lds_b128, overlapped with the
// 8 WMMAs of the previous chunk; retired with s_wait_asynccnt (in-order).
__global__ __launch_bounds__(64) void k_conv_wmma(
    const _Float16* __restrict__ Yh,   // [B*T, 64] f16
    const _Float16* __restrict__ Wpk,  // packed [db][kt(128)][nt(8)][lane][16]
    const float* __restrict__ bias,    // [128]
    float* __restrict__ Cc) {          // [B*T, 128] f32
  __shared__ _Float16 Bsh[2][4096];    // 2 x 8KB
  const int tid  = threadIdx.x;
  const int lane = tid & 31, wv = tid >> 5;
  const int r0   = blockIdx.x * 32 + wv * 16;
  const int bidx = r0 >> 9;            // 512 rows per batch image
  const int t    = (r0 & 511) + (lane & 15);
  const int koff = (lane & 16) ? 8 : 0;

  // each wave copies its half of the 8KB chunk: 8 async b128 per wave
  const unsigned ldsBase[2] = {
      (unsigned)(size_t)(&Bsh[0][wv * 2048]),
      (unsigned)(size_t)(&Bsh[1][wv * 2048])};
  auto issue = [&](int ck, int buf) {
    const _Float16* src = Wpk + (long)ck * 4096 + wv * 2048;
#pragma unroll
    for (int c = 0; c < 8; ++c)
      async_load_b128(ldsBase[buf] + (unsigned)(c * 512 + lane * 16),
                      (const void*)(src + c * 256 + lane * 8));
  };

  v8f acc[8];
#pragma unroll
  for (int j = 0; j < 8; ++j) acc[j] = (v8f){};

  issue(0, 0);
#pragma unroll 1
  for (int ck = 0; ck < 640; ++ck) {
    const int buf = ck & 1;
    const bool more = (ck + 1) < 640;
    if (more) issue(ck + 1, buf ^ 1);        // prefetch next chunk into other buf
    if (more) WAIT_ASYNC(8); else WAIT_ASYNC(0);  // this chunk's 8 loads landed
    __syncthreads();                         // publish buffer to both waves

    const int db = ck >> 7, kt = ck & 127;
    const int sb = bidx + db - 2;
    const int dt = kt >> 1;
    const int ib = ((kt & 1) << 5) + koff;
    const int st = t + dt - 31;
    const bool ok = (sb >= 0) && (sb < Bb_) && (st >= 0) && (st < Tt_);
    const _Float16* asrc = Yh + (ok ? ((long)(sb * Tt_ + st) * 64 + ib) : 0);
    v16h af;
    ((v8h*)&af)[0] = *(const v8h*)(asrc);
    ((v8h*)&af)[1] = *(const v8h*)(asrc + 16);
    if (!ok) af = (v16h){};

#pragma unroll
    for (int nt = 0; nt < 8; ++nt) {
      v16h bf = *(const v16h*)(&Bsh[buf][(nt * 32 + lane) * 16]);  // ds_load_b128 x2
      acc[nt] = __builtin_amdgcn_wmma_f32_16x16x32_f16(false, af, false, bf,
                                                       (short)0, acc[nt], false, false);
    }
    __syncthreads();                         // buffer may be overwritten next iter
  }

  const int nlo = lane & 15;
  const int mo  = (lane & 16) ? 8 : 0;
#pragma unroll
  for (int nt = 0; nt < 8; ++nt) {
    const int nc = nt * 16 + nlo;
    const float bv = bias[nc];
#pragma unroll
    for (int r = 0; r < 8; ++r)
      Cc[(long)(r0 + r + mo) * 128 + nc] = acc[nt][r] + bv;
  }
}

// ---------------- elementwise / softmax kernels -----------------------------
__global__ void k_f32_to_f16(const float* __restrict__ X, _Float16* __restrict__ Y,
                             long n, float scale) {
  long i = (long)blockIdx.x * blockDim.x + threadIdx.x;
  if (i < n) Y[i] = (_Float16)(X[i] * scale);
}

__global__ void k_dup(const float* __restrict__ X, float* __restrict__ Yf,
                      _Float16* __restrict__ Yh, long n) {
  long i = (long)blockIdx.x * blockDim.x + threadIdx.x;
  if (i < n) { float v = X[i]; Yf[i] = v; Yh[i] = (_Float16)v; }
}

__global__ void k_glu(const float* __restrict__ C2, float* __restrict__ Y,
                      _Float16* __restrict__ Yh, long n) {
  long i = (long)blockIdx.x * blockDim.x + threadIdx.x;
  if (i >= n) return;
  long r = i >> 6; int h = (int)(i & 63);
  float a = C2[r * 128 + h], g = C2[r * 128 + 64 + h];
  float v = (a * (1.f / (1.f + __expf(-g))) + Y[i]) * SQ_;
  Y[i] = v; Yh[i] = (_Float16)v;
}

__global__ void k_addscale_h(const float* __restrict__ A, const float* __restrict__ Bx,
                             _Float16* __restrict__ Oh, long n) {
  long i = (long)blockIdx.x * blockDim.x + threadIdx.x;
  if (i < n) Oh[i] = (_Float16)((A[i] + Bx[i]) * SQ_);
}

__global__ __launch_bounds__(256) void k_softmax64(const float* __restrict__ X,
                                                   float* __restrict__ Of,
                                                   _Float16* __restrict__ Oh, int rows) {
  int wv = (int)((blockIdx.x * blockDim.x + threadIdx.x) >> 5);
  int lane = threadIdx.x & 31;
  if (wv >= rows) return;
  float2 v = ((const float2*)(X + (long)wv * 64))[lane];
  float m = fmaxf(v.x, v.y);
  for (int o = 16; o; o >>= 1) m = fmaxf(m, __shfl_xor(m, o, 32));
  float e0 = __expf(v.x - m), e1 = __expf(v.y - m);
  float s = e0 + e1;
  for (int o = 16; o; o >>= 1) s += __shfl_xor(s, o, 32);
  float inv = 1.f / s;
  e0 *= inv; e1 *= inv;
  if (Of) ((float2*)(Of + (long)wv * 64))[lane] = make_float2(e0, e1);
  if (Oh) { _Float16* o = Oh + (long)wv * 64 + 2 * lane;
            o[0] = (_Float16)e0; o[1] = (_Float16)e1; }
}

__global__ __launch_bounds__(256) void k_softmax1024_h(const float* __restrict__ X,
                                                       _Float16* __restrict__ Oh) {
  __shared__ float red[256];
  long row = blockIdx.x; int tid = threadIdx.x;
  float4 v = ((const float4*)(X + row * 1024))[tid];
  float m = fmaxf(fmaxf(v.x, v.y), fmaxf(v.z, v.w));
  red[tid] = m; __syncthreads();
  for (int s = 128; s; s >>= 1) { if (tid < s) red[tid] = fmaxf(red[tid], red[tid + s]); __syncthreads(); }
  m = red[0]; __syncthreads();
  float e0 = __expf(v.x - m), e1 = __expf(v.y - m), e2 = __expf(v.z - m), e3 = __expf(v.w - m);
  red[tid] = e0 + e1 + e2 + e3; __syncthreads();
  for (int s = 128; s; s >>= 1) { if (tid < s) red[tid] += red[tid + s]; __syncthreads(); }
  float inv = 1.f / red[0];
  _Float16* o = Oh + row * 1024 + tid * 4;
  o[0] = (_Float16)(e0 * inv); o[1] = (_Float16)(e1 * inv);
  o[2] = (_Float16)(e2 * inv); o[3] = (_Float16)(e3 * inv);
}

// online softmax over V=32000: one max+sum read pass, one NT-store write pass
__global__ __launch_bounds__(256) void k_softmaxV(float* __restrict__ X, int V) {
  __shared__ float rm[256], rs[256];
  float* xr = X + (long)blockIdx.x * V;
  int tid = threadIdx.x;
  float m = -3.4e38f, s = 0.f;
  for (int i = tid; i < V; i += 256) {
    float x = xr[i];
    float nm = fmaxf(m, x);
    s = s * __expf(m - nm) + __expf(x - nm);
    m = nm;
  }
  rm[tid] = m; rs[tid] = s; __syncthreads();
  for (int o = 128; o; o >>= 1) {
    if (tid < o) {
      float m2 = rm[tid + o], s2 = rs[tid + o];
      float nm = fmaxf(rm[tid], m2);
      rs[tid] = rs[tid] * __expf(rm[tid] - nm) + s2 * __expf(m2 - nm);
      rm[tid] = nm;
    }
    __syncthreads();
  }
  m = rm[0]; float inv = 1.f / rs[0];
  for (int i = tid; i < V; i += 256) {
    float r = __expf(xr[i] - m) * inv;
    __builtin_nontemporal_store(r, xr + i);  // 524MB stream, never re-read
  }
}

// ---------------- host orchestration ----------------------------------------
extern "C" void kernel_launch(void* const* d_in, const int* in_sizes, int n_in,
                              void* d_out, int out_size, void* d_ws, size_t ws_size,
                              hipStream_t stream) {
  const float* input_x = (const float*)d_in[0];
  const float* enc_out = (const float*)d_in[1];
  const float* enc_att = (const float*)d_in[2];
  const float* W1  = (const float*)d_in[3];   const float* b1  = (const float*)d_in[4];
  const float* cW  = (const float*)d_in[5];   const float* cb  = (const float*)d_in[6];
  const float* Wce = (const float*)d_in[7];   const float* bce = (const float*)d_in[8];
  const float* Wec = (const float*)d_in[9];   const float* bec = (const float*)d_in[10];
  const float* W2  = (const float*)d_in[11];  const float* b2  = (const float*)d_in[12];
  const float* W3  = (const float*)d_in[13];  const float* b3  = (const float*)d_in[14];
  float* out = (float*)d_out;

  const long ROWS = (long)Bb_ * Tt_;            // 4096
  const long NX   = ROWS * Ee_;                 // 262144
  const long NL   = ROWS * Ss_;                 // 4194304
  const float ATT_SCALE = 11.313708498984761f;  // E*sqrt(2/E)

  char* w = (char*)d_ws; size_t off = 0;
  auto alloc = [&](size_t bytes) -> char* {
    char* p = w + off; off = (off + bytes + 255) & ~(size_t)255; return p;
  };
  _Float16* hW1p  = (_Float16*)alloc(8 * 1024);
  _Float16* hWcep = (_Float16*)alloc(8 * 1024);
  _Float16* hWecp = (_Float16*)alloc(8 * 1024);
  _Float16* hW2p  = (_Float16*)alloc(8 * 1024);
  _Float16* hW3p  = (_Float16*)alloc(4000L * 1024);
  _Float16* hEAp  = (_Float16*)alloc(8L * 128 * 1024);
  _Float16* hEOp  = (_Float16*)alloc(8L * 128 * 1024);
  _Float16* hCWp  = (_Float16*)alloc(5L * 1024 * 1024);
  _Float16* hx    = (_Float16*)alloc(NX * 2);
  float*    dl1   = (float*)alloc(NX * 4);
  float*    y     = (float*)alloc(NX * 4);
  _Float16* yh    = (_Float16*)alloc(NX * 2);
  float*    cbuf  = (float*)alloc(ROWS * 128 * 4);
  float*    pg    = (float*)alloc(NX * 4);
  _Float16* pgh   = (_Float16*)alloc(NX * 2);
  float*    logits= (float*)alloc(NL * 4);
  _Float16* attnh = (_Float16*)alloc(NL * 2);
  float*    ao    = (float*)alloc(NX * 4);
  _Float16* aoh   = (_Float16*)alloc(NX * 2);
  _Float16* o0h   = (_Float16*)alloc(NX * 2);
  float*    o2    = (float*)alloc(NX * 4);
  _Float16* o2h   = (_Float16*)alloc(NX * 2);
  (void)ws_size; (void)in_sizes; (void)n_in; (void)out_size;

  // ---- pack B operands (per call; O(weights), reused by 32-256 M-tiles) ----
  k_pack_b<<<dim3(2, 4, 1),    32, 0, stream>>>(W1,  0, 64, 1, hW1p,  0, 2, 1);
  k_pack_b<<<dim3(2, 4, 1),    32, 0, stream>>>(Wce, 0, 64, 1, hWcep, 0, 2, 1);
  k_pack_b<<<dim3(2, 4, 1),    32, 0, stream>>>(Wec, 0, 64, 1, hWecp, 0, 2, 1);
  k_pack_b<<<dim3(2, 4, 1),    32, 0, stream>>>(W2,  0, 64, 1, hW2p,  0, 2, 1);
  k_pack_b<<<dim3(2, 2000, 1), 32, 0, stream>>>(W3,  0, Vv_, 1, hW3p, 0, 2, 1);
  k_pack_b<<<dim3(2, 64, 8),   32, 0, stream>>>(enc_att, (long)Ss_ * Ee_, 1, 64,
                                                hEAp, 128, 2, 1);
  k_pack_b<<<dim3(32, 4, 8),   32, 0, stream>>>(enc_out, (long)Ss_ * Ee_, 64, 1,
                                                hEOp, 128, 32, 1);
  k_pack_b<<<dim3(128, 8, 5),  32, 0, stream>>>(cW, 64L * 64 * 128, 128, 1,
                                                hCWp, 1024, 1, 8);

  k_f32_to_f16<<<dim3((unsigned)((NX + 255) / 256)), 256, 0, stream>>>(input_x, hx, NX, 1.f);

  // dl1 = relu(x @ W1 + b1)
  k_wmma_gemm<4><<<dim3(1, 256, 1), 32, 0, stream>>>(hx, 0, 64, hW1p, 0, 2, b1,
                                                     dl1, 0, 64, 64, 1, 1.f);
  k_dup<<<dim3((unsigned)((NX + 255) / 256)), 256, 0, stream>>>(dl1, y, yh, NX);

  for (int l = 0; l < 3; ++l) {
    k_conv_wmma<<<dim3(128), 64, 0, stream>>>(yh, hCWp, cb, cbuf);
    k_glu<<<dim3((unsigned)((NX + 255) / 256)), 256, 0, stream>>>(cbuf, y, yh, NX);
    k_wmma_gemm<4><<<dim3(1, 256, 1), 32, 0, stream>>>(yh, 0, 64, hWcep, 0, 2, bce,
                                                       pg, 0, 64, 64, 0, 1.f);
    k_softmax64<<<dim3(512), 256, 0, stream>>>(pg, nullptr, pgh, (int)ROWS);
    k_wmma_gemm<4><<<dim3(16, 32, 8), 32, 0, stream>>>(pgh, (long)Tt_ * Ee_, 64,
                                                       hEAp, 128L * 512, 2, nullptr,
                                                       logits, (long)Tt_ * Ss_, Ss_,
                                                       64, 0, 1.f);
    k_softmax1024_h<<<dim3((unsigned)ROWS), 256, 0, stream>>>(logits, attnh);
    k_wmma_gemm<4><<<dim3(1, 32, 8), 32, 0, stream>>>(attnh, (long)Tt_ * Ss_, 1024,
                                                      hEOp, 128L * 512, 32, nullptr,
                                                      ao, (long)Tt_ * Ee_, 64,
                                                      1024, 0, ATT_SCALE);
    k_f32_to_f16<<<dim3((unsigned)((NX + 255) / 256)), 256, 0, stream>>>(ao, aoh, NX, 1.f);
    k_wmma_gemm<4><<<dim3(1, 256, 1), 32, 0, stream>>>(aoh, 0, 64, hWecp, 0, 2, bec,
                                                       y, 0, 64, 64, 0, 1.f);
    k_softmax64<<<dim3(512), 256, 0, stream>>>(y, y, yh, (int)ROWS);
  }

  k_addscale_h<<<dim3((unsigned)((NX + 255) / 256)), 256, 0, stream>>>(y, dl1, o0h, NX);
  k_wmma_gemm<4><<<dim3(1, 256, 1), 32, 0, stream>>>(o0h, 0, 64, hW2p, 0, 2, b2,
                                                     o2, 0, 64, 64, 1, 1.f);
  k_f32_to_f16<<<dim3((unsigned)((NX + 255) / 256)), 256, 0, stream>>>(o2, o2h, NX, 1.f);
  k_wmma_gemm<8><<<dim3(250, 256, 1), 32, 0, stream>>>(o2h, 0, 64, hW3p, 0, 2, b3,
                                                       out, 0, Vv_, 64, 1, 1.f);
  k_softmaxV<<<dim3((unsigned)ROWS), 256, 0, stream>>>(out, Vv_);
}
